// AdaAttN_3289944949220
// MI455X (gfx1250) — compile-verified, matrix operand order checked by reading, not run on hardware
//
#include <hip/hip_runtime.h>
#include <hip/hip_bf16.h>

// ---------------------------------------------------------------------------
// AdaAttN fused attention for MI455X (gfx1250, wave32, WMMA bf16 + TDM).
// B=4, Cv=Cqk=512, H=W=64 -> N=4096 pixels per batch.
// ---------------------------------------------------------------------------

typedef __attribute__((ext_vector_type(16))) __bf16 v16bf;
typedef __attribute__((ext_vector_type(8)))  float  v8f;
typedef __attribute__((ext_vector_type(4)))  unsigned int u32x4;
typedef __attribute__((ext_vector_type(8)))  int i32x8;
typedef __attribute__((ext_vector_type(4)))  int i32x4;

union Frag { v16bf v; unsigned short h[16]; };

__device__ __forceinline__ unsigned short f2bf(float f) {
  unsigned u = __builtin_bit_cast(unsigned, f);
  return (unsigned short)((u + 0x7FFFu + ((u >> 16) & 1u)) >> 16);  // RNE
}
__device__ __forceinline__ float bf2f(unsigned short h) {
  unsigned u = ((unsigned)h) << 16;
  return __builtin_bit_cast(float, u);
}

#define NB   4
#define CCH  512
#define HW   4096

// ---------------------------------------------------------------------------
// TDM 2-D tile load Global -> LDS (bf16 elements, data_size code 1 = 2 bytes).
// Descriptor per CDNA5 ISA ch.8: group0 {count=1, lds_addr, global_addr,
// type=2}, group1 {data_size, tensor_dim0/1, tile_dim0/1, dim0_stride}.
// Tracked on TENSORcnt; EXEC ignored; issue from one wave only.
// ---------------------------------------------------------------------------
__device__ __forceinline__ void tdm_load_2d_bf16(
    unsigned lds_off, const void* gptr, unsigned tdim0, unsigned tdim1,
    unsigned tile0, unsigned tile1, unsigned stride0) {
  unsigned long long ga = (unsigned long long)gptr;
  u32x4 g0;
  g0[0] = 1u;                                              // count=1, user D#
  g0[1] = lds_off;                                         // LDS byte address
  g0[2] = (unsigned)ga;                                    // global addr lo
  g0[3] = (unsigned)((ga >> 32) & 0x01FFFFFFu) | (2u << 30);  // hi25 | type=2
  i32x8 g1;
  g1[0] = (int)(1u << 16);                                 // data_size=1 (2B)
  g1[1] = (int)((tdim0 & 0xFFFFu) << 16);                  // tensor_dim0 lo16
  g1[2] = (int)(((tdim0 >> 16) & 0xFFFFu) | ((tdim1 & 0xFFFFu) << 16));
  g1[3] = (int)(((tdim1 >> 16) & 0xFFFFu) | ((tile0 & 0xFFFFu) << 16));
  g1[4] = (int)(tile1 & 0xFFFFu);                          // tile_dim1, dim2=0
  g1[5] = (int)stride0;                                    // dim0_stride lo32
  g1[6] = 0;                                               // hi16 | dim1_stride
  g1[7] = 0;
  i32x4 gz = {0, 0, 0, 0};
#if defined(__clang_major__) && (__clang_major__ >= 23)
  i32x8 gz8 = {0, 0, 0, 0, 0, 0, 0, 0};
  __builtin_amdgcn_tensor_load_to_lds(g0, g1, gz, gz, gz8, 0);
#else
  __builtin_amdgcn_tensor_load_to_lds(g0, g1, gz, gz, 0);
#endif
}

// ---------------------------------------------------------------------------
// Kernel 1: per-channel mean and 1/(std+eps), std with ddof=1 over B*H*W.
// ---------------------------------------------------------------------------
__global__ __launch_bounds__(256) void stats_kernel(
    const float* __restrict__ x, float* __restrict__ mean, float* __restrict__ scale) {
  int c = blockIdx.x;
  float s = 0.f, s2 = 0.f;
  for (int b = 0; b < NB; ++b) {
    const float* p = x + ((long long)b * CCH + c) * HW;
    for (int i = threadIdx.x; i < HW; i += 256) {
      float v = p[i];
      s += v; s2 += v * v;
    }
  }
  __shared__ float r1[256], r2[256];
  r1[threadIdx.x] = s; r2[threadIdx.x] = s2;
  __syncthreads();
  for (int off = 128; off > 0; off >>= 1) {
    if (threadIdx.x < off) {
      r1[threadIdx.x] += r1[threadIdx.x + off];
      r2[threadIdx.x] += r2[threadIdx.x + off];
    }
    __syncthreads();
  }
  if (threadIdx.x == 0) {
    const float n = (float)(NB * HW);
    float m = r1[0] / n;
    float var = fmaxf((r2[0] - n * m * m) / (n - 1.f), 0.f);  // ddof=1
    mean[c]  = m;
    scale[c] = 1.f / (sqrtf(var) + 1e-12f);
  }
}

// ---------------------------------------------------------------------------
// Kernel 2: fused channel-norm + 1x1 conv GEMM, bf16 output.
// Block 256 thr / 8 waves; block tile 32(O) x 64(N); wave tile 16x16.
// Exact-fit float4 staging (no per-element waitcnt serialization).
// ---------------------------------------------------------------------------
__global__ __launch_bounds__(256) void qkv_kernel(
    const float* __restrict__ X, const float* __restrict__ Wt,
    const float* __restrict__ bias, const float* __restrict__ mean,
    const float* __restrict__ scale, int do_norm,
    unsigned short* __restrict__ outb) {
  __shared__ __align__(16) float sW[32 * 36];   // [o][k], 16B-aligned rows
  __shared__ __align__(16) float sX[32 * 68];   // [k][n], 16B-aligned rows

  int b     = blockIdx.z;
  int o_blk = blockIdx.y * 32;
  int n_blk = blockIdx.x * 64;
  int tid = threadIdx.x, lane = tid & 31, w = tid >> 5;
  int o_sub = (w & 1) * 16, n_sub = (w >> 1) * 16;
  int half = lane >> 4, mr = lane & 15, kbase = half * 8;

  // per-thread staging coordinates (exact fit, no loops)
  int woo = tid >> 3, wkk = (tid & 7) * 4;      // W: 32x32 = 256 float4
  int xkk = tid >> 3, xnn = (tid & 7) * 8;      // X: 32x64 = 256 x 2 float4

  v8f acc = {};
  for (int k0 = 0; k0 < CCH; k0 += 32) {
    float4 wv = *(const float4*)(Wt + (long long)(o_blk + woo) * CCH + k0 + wkk);
    *(float4*)(&sW[woo * 36 + wkk]) = wv;

    int c = k0 + xkk;
    float mu = do_norm ? mean[c] : 0.f;
    float sc = do_norm ? scale[c] : 1.f;
    const float4* src = (const float4*)(X + ((long long)b * CCH + c) * HW + n_blk + xnn);
    float4 a = src[0], e = src[1];
    a.x = (a.x - mu) * sc; a.y = (a.y - mu) * sc;
    a.z = (a.z - mu) * sc; a.w = (a.w - mu) * sc;
    e.x = (e.x - mu) * sc; e.y = (e.y - mu) * sc;
    e.z = (e.z - mu) * sc; e.w = (e.w - mu) * sc;
    *(float4*)(&sX[xkk * 68 + xnn])     = a;
    *(float4*)(&sX[xkk * 68 + xnn + 4]) = e;
    __syncthreads();

    // ISA 7.12.2 16-bit A layout: lane -> M=lane&15; K base = 8*(lane>>4);
    // elems j<8 -> K=kbase+j, j>=8 -> K=16+kbase+(j-8).
    Frag A, Bf;
    for (int j = 0; j < 16; ++j) {
      int kk = (j < 8) ? (kbase + j) : (16 + kbase + (j - 8));
      A.h[j]  = f2bf(sW[(o_sub + mr) * 36 + kk]);
      Bf.h[j] = f2bf(sX[kk * 68 + n_sub + mr]);
    }
    acc = __builtin_amdgcn_wmma_f32_16x16x32_bf16(
        false, A.v, false, Bf.v, (short)0, acc, false, false);
    __syncthreads();
  }
  // D layout: vgpr r, lane -> row = r + 8*(lane>>4), col = lane&15
  for (int r = 0; r < 8; ++r) {
    int o = o_blk + o_sub + r + 8 * half;
    int n = n_blk + n_sub + mr;
    float v = acc[r] + bias[o];
    outb[((long long)b * CCH + o) * HW + n] = f2bf(v);
  }
}

// ---------------------------------------------------------------------------
// Kernel 3: fused flash-attention with weighted mean + variance.
// One (batch, 16-query) tile per 8-wave block; keys streamed in 32-wide
// tiles, TDM double-buffered Global->LDS. Accumulators [Cv=512 x 16q] for
// M and E[V^2] distributed across waves (64 acc VGPRs/lane).
// ---------------------------------------------------------------------------
__global__ __launch_bounds__(256) void attn_kernel(
    const unsigned short* __restrict__ Qb, const unsigned short* __restrict__ Kb,
    const unsigned short* __restrict__ Vb, const float* __restrict__ F_c,
    const float* __restrict__ fc_mean, const float* __restrict__ fc_scale,
    float* __restrict__ out) {
  extern __shared__ char smem[];
  // LDS map (byte offsets double as TDM lds_addr)
  const unsigned OFF_Q  = 0;                       // [c=512][q=16] bf16  16 KB
  const unsigned OFF_K0 = OFF_Q  + CCH * 16 * 2;   // [c=512][m=32] bf16  32 KB
  const unsigned OFF_V0 = OFF_K0 + CCH * 32 * 2;
  const unsigned OFF_K1 = OFF_V0 + CCH * 32 * 2;   // ping-pong buffers
  const unsigned OFF_V1 = OFF_K1 + CCH * 32 * 2;
  const unsigned OFF_S  = OFF_V1 + CCH * 32 * 2;   // float[16*33]
  const unsigned OFF_P  = OFF_S + 16 * 33 * 4;     // bf16 [m=32][q=16]
  const unsigned OFF_ST = OFF_P + 32 * 16 * 2;     // sMax/sSum/sAlpha

  unsigned short* sQ = (unsigned short*)(smem + OFF_Q);
  float*          sS = (float*)(smem + OFF_S);
  unsigned short* sP = (unsigned short*)(smem + OFF_P);
  float* sMax   = (float*)(smem + OFF_ST);
  float* sSum   = sMax + 16;
  float* sAlpha = sSum + 16;

  int b  = blockIdx.y;
  int n0 = blockIdx.x * 16;
  int tid = threadIdx.x, lane = tid & 31, w = tid >> 5;
  int half = lane >> 4, mr = lane & 15, kbase = half * 8;

  const unsigned short* Kg = Kb + (long long)b * CCH * HW;
  const unsigned short* Vg = Vb + (long long)b * CCH * HW;

  // Prologue: TDM Q tile + first K/V tiles (wave 0 only; TDM ignores EXEC).
  if (w == 0) {
    tdm_load_2d_bf16(OFF_Q,  Qb + (long long)b * CCH * HW + n0, HW, CCH, 16, CCH, HW);
    tdm_load_2d_bf16(OFF_K0, Kg, HW, CCH, 32, CCH, HW);
    tdm_load_2d_bf16(OFF_V0, Vg, HW, CCH, 32, CCH, HW);
    __builtin_amdgcn_s_wait_tensorcnt(0);
  }
  if (tid < 16) { sMax[tid] = -3.0e38f; sSum[tid] = 0.f; }

  v8f accM[4], accS[4];
  for (int t = 0; t < 4; ++t) { accM[t] = (v8f){}; accS[t] = (v8f){}; }
  __syncthreads();

  for (int it = 0; it < HW / 32; ++it) {
    int cur = it & 1;
    int m0 = it * 32;
    // Prefetch next K/V tiles into the other buffer (overlaps with compute).
    if (w == 0 && it + 1 < HW / 32) {
      tdm_load_2d_bf16(cur ? OFF_K0 : OFF_K1, Kg + m0 + 32, HW, CCH, 32, CCH, HW);
      tdm_load_2d_bf16(cur ? OFF_V0 : OFF_V1, Vg + m0 + 32, HW, CCH, 32, CCH, HW);
    }
    unsigned short* sK = (unsigned short*)(smem + (cur ? OFF_K1 : OFF_K0));
    unsigned short* sV = (unsigned short*)(smem + (cur ? OFF_V1 : OFF_V0));

    for (int i = tid; i < 16 * 32; i += 256)
      sS[(i >> 5) * 33 + (i & 31)] = 0.f;
    __syncthreads();

    // Scores S[16q x 32m] = Q^T K : wave w owns C-slice [w*64, w*64+64).
    for (int hmm = 0; hmm < 2; ++hmm) {
      v8f acc = {};
      for (int kk0 = w * 64; kk0 < w * 64 + 64; kk0 += 32) {
        Frag A, Bf;
        for (int j = 0; j < 16; ++j) {
          int kk = kk0 + ((j < 8) ? (kbase + j) : (16 + kbase + (j - 8)));
          A.h[j]  = sQ[kk * 16 + mr];              // A[q][c]
          Bf.h[j] = sK[kk * 32 + hmm * 16 + mr];   // B[c][m]
        }
        acc = __builtin_amdgcn_wmma_f32_16x16x32_bf16(
            false, A.v, false, Bf.v, (short)0, acc, false, false);
      }
      for (int r = 0; r < 8; ++r) {
        int q  = r + 8 * half;
        int mm = hmm * 16 + mr;
        atomicAdd(&sS[q * 33 + mm], acc[r]);       // ds_add_f32 reduction
      }
    }
    __syncthreads();

    // Online softmax update (lanes 0..15 of wave 0, one query row each).
    if (tid < 16) {
      int q = tid;
      float nm = sMax[q];
      for (int mm = 0; mm < 32; ++mm) nm = fmaxf(nm, sS[q * 33 + mm]);
      float alpha = __expf(sMax[q] - nm);
      float s = sSum[q] * alpha;
      for (int mm = 0; mm < 32; ++mm) {
        float p = __expf(sS[q * 33 + mm] - nm);
        s += p;
        sP[mm * 16 + q] = f2bf(p);                 // B layout [m][q]
      }
      sMax[q] = nm; sSum[q] = s; sAlpha[q] = alpha;
    }
    __syncthreads();

    // Rescale running accumulators, add V*P and V^2*P for our 4 Cv-tiles.
    float alpha = sAlpha[mr];                      // col of D tile == q
    Frag Bp;
    for (int j = 0; j < 16; ++j) {
      int kk = (j < 8) ? (kbase + j) : (16 + kbase + (j - 8));
      Bp.h[j] = sP[kk * 16 + mr];
    }
    for (int t = 0; t < 4; ++t) {
      int v0 = (w * 4 + t) * 16;
      Frag A1, A2;
      for (int j = 0; j < 16; ++j) {
        int kk = (j < 8) ? (kbase + j) : (16 + kbase + (j - 8));
        unsigned short hv = sV[(v0 + mr) * 32 + kk];
        A1.h[j] = hv;
        float fv = bf2f(hv);
        A2.h[j] = f2bf(fv * fv);                   // V^2 fragment on the fly
      }
      for (int r = 0; r < 8; ++r) { accM[t][r] *= alpha; accS[t][r] *= alpha; }
      accM[t] = __builtin_amdgcn_wmma_f32_16x16x32_bf16(
          false, A1.v, false, Bp.v, (short)0, accM[t], false, false);
      accS[t] = __builtin_amdgcn_wmma_f32_16x16x32_bf16(
          false, A2.v, false, Bp.v, (short)0, accS[t], false, false);
    }
    // Ensure next-tile DMA landed before anyone flips buffers.
    if (w == 0 && it + 1 < HW / 32) __builtin_amdgcn_s_wait_tensorcnt(0);
    __syncthreads();
  }

  // Epilogue: M, var -> S, fuse channel_norm(F_c), write fp32 output.
  float inv = 1.f / sSum[mr];
  for (int t = 0; t < 4; ++t) {
    int v0 = (w * 4 + t) * 16;
    for (int r = 0; r < 8; ++r) {
      int v = v0 + r + 8 * half;
      int n = n0 + mr;
      float M  = accM[t][r] * inv;
      float E2 = accS[t][r] * inv;
      float var = fmaxf(E2 - M * M, 0.f);
      float Sd = sqrtf(var + 1e-8f);
      if (!(Sd == Sd)) Sd = 0.f;                   // nan_to_num
      long long idx = ((long long)b * CCH + v) * HW + n;
      float fc = (F_c[idx] - fc_mean[v]) * fc_scale[v];
      out[idx] = Sd * fc + M;
    }
  }
}

// ---------------------------------------------------------------------------
// Host-side launch.
// ---------------------------------------------------------------------------
extern "C" void kernel_launch(void* const* d_in, const int* in_sizes, int n_in,
                              void* d_out, int out_size, void* d_ws, size_t ws_size,
                              hipStream_t stream) {
  const float* F_c  = (const float*)d_in[0];
  const float* F_s  = (const float*)d_in[1];
  const float* F_cp = (const float*)d_in[2];
  const float* F_sp = (const float*)d_in[3];
  const float* Wf   = (const float*)d_in[4];
  const float* bf   = (const float*)d_in[5];
  const float* Wg   = (const float*)d_in[6];
  const float* bg   = (const float*)d_in[7];
  const float* Wh   = (const float*)d_in[8];
  const float* bh   = (const float*)d_in[9];
  float* out = (float*)d_out;

  // Workspace: 6x512 floats of stats, then bf16 Q, K, V tensors (16 MB each).
  char* ws = (char*)d_ws;
  float* mQ = (float*)ws;
  float* sQ = mQ + CCH;
  float* mK = sQ + CCH;
  float* sK = mK + CCH;
  float* mC = sK + CCH;
  float* sC = mC + CCH;
  const size_t qkv_bytes = (size_t)NB * CCH * HW * sizeof(unsigned short);
  unsigned short* Qb = (unsigned short*)(ws + 16384);
  unsigned short* Kb = (unsigned short*)((char*)Qb + qkv_bytes);
  unsigned short* Vb = (unsigned short*)((char*)Kb + qkv_bytes);

  // 1) channel statistics
  stats_kernel<<<CCH, 256, 0, stream>>>(F_cp, mQ, sQ);
  stats_kernel<<<CCH, 256, 0, stream>>>(F_sp, mK, sK);
  stats_kernel<<<CCH, 256, 0, stream>>>(F_c,  mC, sC);

  // 2) Q = Wf*norm(F_cp)+bf ; K = Wg*norm(F_sp)+bg ; V = Wh*F_s+bh (bf16)
  dim3 ggrid(HW / 64, CCH / 32, NB);
  qkv_kernel<<<ggrid, 256, 0, stream>>>(F_cp, Wf, bf, mQ, sQ, 1, Qb);
  qkv_kernel<<<ggrid, 256, 0, stream>>>(F_sp, Wg, bg, mK, sK, 1, Kb);
  qkv_kernel<<<ggrid, 256, 0, stream>>>(F_s,  Wh, bh, mQ, sQ, 0, Vb);

  // 3) fused attention; dynamic LDS: Q + 2x(K,V) ping-pong + softmax state
  size_t smem = (size_t)CCH * 16 * 2        // sQ
              + (size_t)4 * CCH * 32 * 2    // K/V double buffers
              + (size_t)(16 * 33) * 4       // sS
              + (size_t)(32 * 16) * 2       // sP
              + (size_t)48 * 4;             // sMax/sSum/sAlpha
  attn_kernel<<<dim3(HW / 16, NB), 256, smem, stream>>>(Qb, Kb, Vb, F_c, mC, sC, out);
}
